// CrossAttentionLayer_70944269795485
// MI455X (gfx1250) — matrix operand order repaired and checked by reading, not compiled, not run on hardware
//
#include <hip/hip_runtime.h>
#include <hip/hip_bf16.h>
#include <math.h>

#define B_  4
#define SQ_ 512
#define SK_ 2048
#define H_  1024
#define NH_ 16
#define HD_ 64

typedef __attribute__((ext_vector_type(16))) _Float16 v16h;
typedef __attribute__((ext_vector_type(8)))  float    v8f;

// ---------------------------------------------------------------------------
// CDNA5 async memory->LDS copy (16B per lane), tracked by ASYNCcnt.
// LDS operand: low 32 bits of a generic pointer to __shared__ are the
// workgroup-relative LDS byte offset.
// ---------------------------------------------------------------------------
__device__ __forceinline__ void async_ld_b128(void* lds, const void* gptr) {
    unsigned int       l = (unsigned int)(unsigned long long)(uintptr_t)lds;
    unsigned long long g = (unsigned long long)(uintptr_t)gptr;
    asm volatile("global_load_async_to_lds_b128 %0, %1, off"
                 :: "v"(l), "v"(g) : "memory");
}
__device__ __forceinline__ void wait_async0() {
    asm volatile("s_wait_asynccnt 0x0" ::: "memory");
}

// ---------------------------------------------------------------------------
// Kernel 1: LayerNorm (f32) + cast to f16.  One block per row of H=1024.
// ---------------------------------------------------------------------------
__global__ void ln_cast_kernel(const float* __restrict__ x,
                               const float* __restrict__ g,
                               const float* __restrict__ bb,
                               _Float16* __restrict__ out) {
    __shared__ float red[256];
    const int row = blockIdx.x;
    const int tid = threadIdx.x;
    const float* xr = x + (size_t)row * H_;
    float v[4];
    float s = 0.f;
#pragma unroll
    for (int j = 0; j < 4; ++j) { v[j] = xr[tid + 256 * j]; s += v[j]; }
    red[tid] = s; __syncthreads();
    for (int off = 128; off > 0; off >>= 1) {
        if (tid < off) red[tid] += red[tid + off];
        __syncthreads();
    }
    const float mu = red[0] * (1.0f / H_);
    __syncthreads();
    s = 0.f;
#pragma unroll
    for (int j = 0; j < 4; ++j) { float d = v[j] - mu; s += d * d; }
    red[tid] = s; __syncthreads();
    for (int off = 128; off > 0; off >>= 1) {
        if (tid < off) red[tid] += red[tid + off];
        __syncthreads();
    }
    const float rstd = rsqrtf(red[0] * (1.0f / H_) + 1e-12f);
    _Float16* orow = out + (size_t)row * H_;
#pragma unroll
    for (int j = 0; j < 4; ++j) {
        int i = tid + 256 * j;
        orow[i] = (_Float16)((v[j] - mu) * rstd * g[i] + bb[i]);
    }
}

// ---------------------------------------------------------------------------
// Kernel 2: f32 -> f16 cast, 8 elements/thread (b128 in, b128 out).
// ---------------------------------------------------------------------------
__global__ void cast_f16_kernel(const float* __restrict__ in,
                                _Float16* __restrict__ out, int n) {
    for (int i8 = blockIdx.x * blockDim.x + threadIdx.x; i8 < (n >> 3);
         i8 += gridDim.x * blockDim.x) {
        const float* src = in + (size_t)i8 * 8;
        _Float16 tmp[8];
#pragma unroll
        for (int j = 0; j < 8; ++j) tmp[j] = (_Float16)src[j];
        *(uint4*)&out[(size_t)i8 * 8] = *(const uint4*)tmp;
    }
}

// ---------------------------------------------------------------------------
// Kernel 3: y = x @ W^T + bias, f16 in, f32 acc.  Block tile 64x64, 8 waves
// (2m x 4n), two 16x16 WMMA tiles per wave.  Tiles staged to LDS with
// GLOBAL_LOAD_ASYNC_TO_LDS_B128, double-buffered: one barrier per K-step.
// sched_barrier(0) pins the WMMAs before the next iteration's asynccnt wait
// so each tile's fills overlap a full compute phase.
// ---------------------------------------------------------------------------
__global__ void gemm_wmma_kernel(const _Float16* __restrict__ A,   // [M][1024]
                                 const _Float16* __restrict__ W,   // [1024][1024]
                                 const float* __restrict__ bias,   // [1024]
                                 _Float16* __restrict__ out,       // [B][NH][seq][64]
                                 int lgSeq) {
    __shared__ _Float16 sA[2][64][40];   // 40-half stride (20 dwords): conflict-free
    __shared__ _Float16 sW[2][64][40];
    const int seqLen = 1 << lgSeq;
    const int nb = blockIdx.x & 15;          // H/64 = 16 column blocks
    const int mb = blockIdx.x >> 4;
    const int m0 = mb * 64, n0 = nb * 64;
    const int tid  = threadIdx.x;
    const int wid  = tid >> 5, lane = tid & 31;
    const int wm   = wid & 1,  wn   = wid >> 1;
    const int r    = lane & 15, hsel = lane >> 4;
    const int lrow = tid >> 2, lcol = (tid & 3) * 8;   // 16B per thread per tile
    const _Float16* gA = &A[(size_t)(m0 + lrow) * H_ + lcol];
    const _Float16* gW = &W[(size_t)(n0 + lrow) * H_ + lcol];

    v8f acc0 = {}, acc1 = {};
    int p = 0;
    async_ld_b128(&sA[0][lrow][lcol], gA);             // stage K-tile 0
    async_ld_b128(&sW[0][lrow][lcol], gW);
    for (int kk = 0; kk < H_; kk += 32) {
        wait_async0();                                 // own async fills done
        __syncthreads();                               // tile kk published
        if (kk + 32 < H_) {                            // fill other buffer
            async_ld_b128(&sA[p ^ 1][lrow][lcol], gA + kk + 32);
            async_ld_b128(&sW[p ^ 1][lrow][lcol], gW + kk + 32);
        }
        v16h a0, a1, bf;
#pragma unroll
        for (int j = 0; j < 8; ++j) {
            // 16-bit A layout: vgpr j holds K pair (j<4 ? 2j : 8+2j) + 8*(lane>=16)
            int k0 = (j < 4 ? 2 * j : 8 + 2 * j) + 8 * hsel;
            a0[2 * j]     = sA[p][32 * wm + r][k0];
            a0[2 * j + 1] = sA[p][32 * wm + r][k0 + 1];
            a1[2 * j]     = sA[p][32 * wm + 16 + r][k0];
            a1[2 * j + 1] = sA[p][32 * wm + 16 + r][k0 + 1];
            // 16-bit B layout: same K order as A, col = lane&15
            int kb = (j < 4 ? 2 * j : 8 + 2 * j) + 8 * hsel;
            bf[2 * j]     = sW[p][16 * wn + r][kb];
            bf[2 * j + 1] = sW[p][16 * wn + r][kb + 1];
        }
        acc0 = __builtin_amdgcn_wmma_f32_16x16x32_f16(false, a0, false, bf, (short)0, acc0, false, false);
        acc1 = __builtin_amdgcn_wmma_f32_16x16x32_f16(false, a1, false, bf, (short)0, acc1, false, false);
        __builtin_amdgcn_sched_barrier(0);             // keep WMMAs before next wait
        p ^= 1;
    }
    const int n  = n0 + 16 * wn + r;
    const float bv = bias[n];
    const int hh = n >> 6, dd = n & 63;
#pragma unroll
    for (int i = 0; i < 8; ++i) {
        int m  = m0 + 32 * wm + i + 8 * hsel;       // C layout: M = vgpr + 8*(lane>=16)
        int bq = m >> lgSeq, pa = m & (seqLen - 1);
        out[((size_t)(bq * NH_ + hh) << lgSeq) * HD_ + (size_t)pa * HD_ + dd] = (_Float16)(acc0[i] + bv);
        m += 16; bq = m >> lgSeq; pa = m & (seqLen - 1);
        out[((size_t)(bq * NH_ + hh) << lgSeq) * HD_ + (size_t)pa * HD_ + dd] = (_Float16)(acc1[i] + bv);
    }
}

// ---------------------------------------------------------------------------
// Kernel 4 (fused attention): block = (b, h, 32 q-rows).
// Phase 0: scores = (Q K^T)/8 with WMMA; f32 -> d_out, f16 copy -> LDS,
//          per-lane row-max in registers, reduced via 8KB LDS array.
//          K stream prefetched 2 tiles ahead (global_prefetch).
// Phase 1: softmax entirely from LDS (no global read-back of scores).
// Phase 2: ctx = P @ V with WMMA; V tiles async-copied, double-buffered.
// LDS: sP 32x2056 f16 + sV 2x32x72 f16 + red 1KB + redM 8KB = 150,016 B.
// ---------------------------------------------------------------------------
__global__ void attn_fused_kernel(const _Float16* __restrict__ q16,   // [BH][SQ][64]
                                  const _Float16* __restrict__ k16,   // [BH][SK][64]
                                  const _Float16* __restrict__ enc16, // [B][SK][H]
                                  const int* __restrict__ mask,       // [B][SK]
                                  float* __restrict__ scores,         // [BH][SQ][SK]
                                  float* __restrict__ ctx) {          // [B][SQ][H]
    extern __shared__ _Float16 smem[];
    const int PSTRIDE = SK_ + 8;               // 2056 halves = 1028 dwords (≡4 mod 64 banks)
    const int VBUF    = 32 * 72;               // one V buffer: 32 keys x 64 d (+8 pad)
    _Float16* sP = smem;                       // 32 * 2056
    _Float16* sV = smem + 32 * PSTRIDE;        // 2 * VBUF (double buffer)
    float* red   = (float*)(smem + 32 * PSTRIDE + 2 * VBUF);   // 256 f32
    float* redM  = red + 256;                  // 32 x 64 f32 row-max partials

    const int qb = blockIdx.x & 15;            // SQ/32 = 16
    const int bh = blockIdx.x >> 4;
    const int b = bh / NH_, h = bh % NH_;
    const int q0 = qb * 32;
    const int tid = threadIdx.x;
    const int wid = tid >> 5, lane = tid & 31;
    const int r = lane & 15, hsel = lane >> 4;

    // ---- phase 0: WMMA scores, write f32 out + f16 to LDS, track row max ----
    {
        const int qsub = wid >> 2;             // 0..1 : which 16-row group
        const int kq   = wid & 3;              // 0..3 : key sub-tile within 64
        const _Float16* qrow = q16 + ((size_t)bh * SQ_ + q0 + qsub * 16 + r) * HD_;
        v16h aq0, aq1;
#pragma unroll
        for (int j = 0; j < 8; ++j) {
            int k0 = (j < 4 ? 2 * j : 8 + 2 * j) + 8 * hsel;
            aq0[2 * j]     = qrow[k0];
            aq0[2 * j + 1] = qrow[k0 + 1];
            aq1[2 * j]     = qrow[k0 + 32];
            aq1[2 * j + 1] = qrow[k0 + 33];
        }
        float vmax[8];
#pragma unroll
        for (int i = 0; i < 8; ++i) vmax[i] = -INFINITY;

        const _Float16* kbase = k16 + ((size_t)bh * SK_ + kq * 16 + r) * HD_;
        for (int kt = 0; kt < SK_ / 64; ++kt) {        // 32 key tiles per wave
            const int key0 = kt * 64 + kq * 16;
            const _Float16* kcol = kbase + (size_t)kt * 64 * HD_;
            if (kt + 2 < SK_ / 64)                     // hide HBM latency on K stream
                __builtin_prefetch(kcol + 2 * 64 * HD_, 0, 1);
            v16h b0, b1;
#pragma unroll
            for (int j = 0; j < 8; ++j) {
                int kb = (j < 4 ? 2 * j : 8 + 2 * j) + 8 * hsel;
                b0[2 * j]     = kcol[kb];
                b0[2 * j + 1] = kcol[kb + 1];
                b1[2 * j]     = kcol[kb + 32];
                b1[2 * j + 1] = kcol[kb + 33];
            }
            v8f acc = {};
            acc = __builtin_amdgcn_wmma_f32_16x16x32_f16(false, aq0, false, b0, (short)0, acc, false, false);
            acc = __builtin_amdgcn_wmma_f32_16x16x32_f16(false, aq1, false, b1, (short)0, acc, false, false);
            const int key = key0 + r;                  // C layout: N = lane&15
            const int mv  = mask[b * SK_ + key];
#pragma unroll
            for (int i = 0; i < 8; ++i) {
                int qr  = qsub * 16 + i + 8 * hsel;    // row within the 32-row block
                float val = mv ? acc[i] * 0.125f : -INFINITY;
                scores[((size_t)bh * SQ_ + q0 + qr) * SK_ + key] = val;
                sP[qr * PSTRIDE + key] = (_Float16)val;
                vmax[i] = fmaxf(vmax[i], val);
            }
        }
#pragma unroll
        for (int i = 0; i < 8; ++i)
            redM[(qsub * 16 + i + 8 * hsel) * 64 + kq * 16 + r] = vmax[i];
    }
    __syncthreads();

    // ---- phase 1: softmax of 32 rows x 2048, all operands in LDS ----
    const int row = tid >> 3, sub = tid & 7;
    float m = -INFINITY;
#pragma unroll
    for (int t = 0; t < 8; ++t) m = fmaxf(m, redM[row * 64 + sub + 8 * t]);
    red[tid] = m; __syncthreads();
    if (sub < 4) red[tid] = fmaxf(red[tid], red[tid + 4]); __syncthreads();
    if (sub < 2) red[tid] = fmaxf(red[tid], red[tid + 2]); __syncthreads();
    if (sub < 1) red[tid] = fmaxf(red[tid], red[tid + 1]); __syncthreads();
    const float rmax = red[row * 8];
    __syncthreads();
    float s = 0.f;
    for (int jj = 0; jj < SK_ / 8; ++jj) {
        int col = sub + 8 * jj;
        float e = __expf((float)sP[row * PSTRIDE + col] - rmax);  // exp(-inf)=0 for masked
        s += e;
        sP[row * PSTRIDE + col] = (_Float16)e;
    }
    red[tid] = s; __syncthreads();
    if (sub < 4) red[tid] += red[tid + 4]; __syncthreads();
    if (sub < 2) red[tid] += red[tid + 2]; __syncthreads();
    if (sub < 1) red[tid] += red[tid + 1]; __syncthreads();
    const float rinv = 1.0f / red[row * 8];
    __syncthreads();
    for (int jj = 0; jj < SK_ / 8; ++jj) {
        int col = sub + 8 * jj;
        sP[row * PSTRIDE + col] = (_Float16)((float)sP[row * PSTRIDE + col] * rinv);
    }

    // ---- phase 2: ctx = P @ V  (V tiles async-copied, double-buffered) ----
    const int qsub = wid >> 2, dsub = wid & 3;
    const int vrow = tid >> 3, vcol = (tid & 7) * 8;
    const _Float16* gV = &enc16[((size_t)b * SK_ + vrow) * H_ + h * HD_ + vcol];
    int p = 0;
    async_ld_b128(&sV[vrow * 72 + vcol], gV);          // preload key block 0
    v8f acc = {};
    for (int key0 = 0; key0 < SK_; key0 += 32) {
        wait_async0();                                 // own V fills done
        __syncthreads();                               // publish sV[p] (+ phase-1 sP)
        if (key0 + 32 < SK_) {                         // prefetch next tile
            async_ld_b128(&sV[(p ^ 1) * VBUF + vrow * 72 + vcol],
                          gV + (size_t)(key0 + 32) * H_);
        }
        const _Float16* sVp = sV + p * VBUF;
        v16h a, bf;
#pragma unroll
        for (int j = 0; j < 8; ++j) {
            int k0 = (j < 4 ? 2 * j : 8 + 2 * j) + 8 * hsel;
            a[2 * j]     = sP[(qsub * 16 + r) * PSTRIDE + key0 + k0];
            a[2 * j + 1] = sP[(qsub * 16 + r) * PSTRIDE + key0 + k0 + 1];
            int kb = (j < 4 ? 2 * j : 8 + 2 * j) + 8 * hsel;
            bf[2 * j]     = sVp[kb * 72 + dsub * 16 + r];
            bf[2 * j + 1] = sVp[(kb + 1) * 72 + dsub * 16 + r];
        }
        acc = __builtin_amdgcn_wmma_f32_16x16x32_f16(false, a, false, bf, (short)0, acc, false, false);
        __builtin_amdgcn_sched_barrier(0);             // keep WMMA before next wait
        p ^= 1;
    }
    const int d = dsub * 16 + r;
#pragma unroll
    for (int i = 0; i < 8; ++i) {
        int q = q0 + qsub * 16 + i + 8 * hsel;
        ctx[((size_t)b * SQ_ + q) * H_ + h * HD_ + d] = acc[i];
    }
}

// ---------------------------------------------------------------------------
// Launch
// ---------------------------------------------------------------------------
extern "C" void kernel_launch(void* const* d_in, const int* in_sizes, int n_in,
                              void* d_out, int out_size, void* d_ws, size_t ws_size,
                              hipStream_t stream) {
    (void)in_sizes; (void)n_in; (void)out_size; (void)ws_size;
    const float* hs   = (const float*)d_in[0];
    const float* enc  = (const float*)d_in[1];
    const int*   mask = (const int*)d_in[2];
    const float* qw   = (const float*)d_in[3];
    const float* qb   = (const float*)d_in[4];
    const float* kw   = (const float*)d_in[5];
    const float* kb   = (const float*)d_in[6];
    const float* lng  = (const float*)d_in[7];
    const float* lnb  = (const float*)d_in[8];

    float* out    = (float*)d_out;
    float* ctx    = out;                                   // [B][SQ][H]
    float* scores = out + (size_t)B_ * SQ_ * H_;           // [B][NH][SQ][SK]

    // workspace (f16), ~46 MB total
    _Float16* w     = (_Float16*)d_ws;
    _Float16* hs16  = w;                                   // 2,097,152
    _Float16* enc16 = hs16  + (size_t)B_ * SQ_ * H_;       // 8,388,608
    _Float16* qw16  = enc16 + (size_t)B_ * SK_ * H_;       // 1,048,576
    _Float16* kw16  = qw16  + (size_t)H_ * H_;             // 1,048,576
    _Float16* q16   = kw16  + (size_t)H_ * H_;             // 2,097,152
    _Float16* k16   = q16   + (size_t)B_ * SQ_ * H_;       // 8,388,608

    ln_cast_kernel<<<B_ * SQ_, 256, 0, stream>>>(hs, lng, lnb, hs16);
    cast_f16_kernel<<<1024, 256, 0, stream>>>(enc, enc16, B_ * SK_ * H_);
    cast_f16_kernel<<<256, 256, 0, stream>>>(qw, qw16, H_ * H_);
    cast_f16_kernel<<<256, 256, 0, stream>>>(kw, kw16, H_ * H_);

    gemm_wmma_kernel<<<(B_ * SQ_ / 64) * (H_ / 64), 256, 0, stream>>>(hs16, qw16, qb, q16, 9);   // seq=512
    gemm_wmma_kernel<<<(B_ * SK_ / 64) * (H_ / 64), 256, 0, stream>>>(enc16, kw16, kb, k16, 11); // seq=2048

    const size_t lds_bytes = (size_t)(32 * (SK_ + 8) + 2 * 32 * 72) * sizeof(_Float16)
                           + (256 + 32 * 64) * sizeof(float);  // 150,016 B
    attn_fused_kernel<<<(B_ * NH_) * (SQ_ / 32), 256, lds_bytes, stream>>>(
        q16, k16, enc16, mask, scores, ctx);
}
